// Seq2SeqForecaster_19292993093921
// MI455X (gfx1250) — matrix operand order repaired and checked by reading, not compiled, not run on hardware
//
#include <hip/hip_runtime.h>

typedef __attribute__((ext_vector_type(16))) _Float16 v16h;
typedef __attribute__((ext_vector_type(8)))  _Float16 v8h;
typedef __attribute__((ext_vector_type(8)))  float    v8f;

#define HD 256
#define BN 256
#define LN 512
#define PL 96

// ---------------- workspace layout (bytes) ----------------
static constexpr size_t OFF_WH0  = 0;                       // 768*256 f16
static constexpr size_t OFF_WI1  = OFF_WH0 + 393216;
static constexpr size_t OFF_WH1  = OFF_WI1 + 393216;
static constexpr size_t OFF_WDH  = OFF_WH1 + 393216;
static constexpr size_t OFF_XT   = OFF_WDH + 393216;        // 512*256 f32 (x transposed)
static constexpr size_t OFF_H0A  = OFF_XT  + 524288;        // 256*256 f16 ping/pong
static constexpr size_t OFF_H0B  = OFF_H0A + 131072;
static constexpr size_t OFF_H1A  = OFF_H0B + 131072;
static constexpr size_t OFF_H1B  = OFF_H1A + 131072;
static constexpr size_t OFF_PRED = OFF_H1B + 131072;        // 256 f32
static constexpr size_t OFF_BAR  = OFF_PRED + 1024;         // 1 u32
static constexpr size_t OFF_YS0  = 4194304;                 // 512*256*256 f16 = 64 MB

// ---------------- fragment loads (CDNA5 wave32 WMMA layouts) ----------------
union FragU { v16h v; v8h h2[2]; };

// A-matrix 16x32 f16: lane m = lane&15; K-half = 16*(lane>>4); 16 contiguous halfs
__device__ __forceinline__ v16h loadA(const _Float16* __restrict__ arow, int kk, int lane) {
  const _Float16* p = arow + kk * 32 + ((lane >> 4) << 4);
  FragU u;
  u.h2[0] = *(const v8h*)(p);
  u.h2[1] = *(const v8h*)(p + 8);
  return u.v;
}

// B-matrix 32x16 f16: lane n = lane&15; K-half = 16*(lane>>4); 16 contiguous halfs of W row n
__device__ __forceinline__ v16h loadB(const _Float16* __restrict__ wrow, int kk, int lane) {
  const _Float16* p = wrow + kk * 32 + ((lane >> 4) << 4);
  FragU u;
  u.h2[0] = *(const v8h*)(p);
  u.h2[1] = *(const v8h*)(p + 8);
  return u.v;
}

// Accumulate 3 gate tiles (r,z,n) of  A[16,256] @ W[768,256]^T  for tile (mbase, jbase).
// A-fragments shared across the 3 gates -> 8 A-loads, 24 B-loads, 24 WMMAs.
__device__ __forceinline__ void gemm3(const _Float16* __restrict__ A,
                                      const _Float16* __restrict__ W,
                                      int mbase, int jbase, int lane,
                                      v8f& c0, v8f& c1, v8f& c2) {
  const _Float16* arow = A + (size_t)(mbase + (lane & 15)) * HD;
  const _Float16* w0   = W + (size_t)(jbase + (lane & 15)) * HD;   // gate r rows
  const _Float16* w1   = w0 + (size_t)HD * HD;                     // gate z rows
  const _Float16* w2   = w1 + (size_t)HD * HD;                     // gate n rows
#pragma unroll
  for (int kk = 0; kk < 8; ++kk) {
    v16h a = loadA(arow, kk, lane);
    c0 = __builtin_amdgcn_wmma_f32_16x16x32_f16(false, a, false, loadB(w0, kk, lane), (short)0, c0, false, false);
    c1 = __builtin_amdgcn_wmma_f32_16x16x32_f16(false, a, false, loadB(w1, kk, lane), (short)0, c1, false, false);
    c2 = __builtin_amdgcn_wmma_f32_16x16x32_f16(false, a, false, loadB(w2, kk, lane), (short)0, c2, false, false);
  }
}

__device__ __forceinline__ float sigf(float x) { return 1.0f / (1.0f + __expf(-x)); }
__device__ __forceinline__ float tanhf_fast(float x) {
  x = fminf(15.0f, fmaxf(-15.0f, x));
  float e = __expf(-2.0f * x);
  return (1.0f - e) / (1.0f + e);
}

// Monotonic-counter grid barrier (16 blocks resident; counter zeroed by prep kernel each launch)
__device__ __forceinline__ void grid_sync(unsigned* bar, unsigned target) {
  __threadfence();
  __syncthreads();
  if (threadIdx.x == 0) {
    atomicAdd(bar, 1u);
    while (*((volatile unsigned*)bar) < target) { __builtin_amdgcn_s_sleep(2); }
  }
  __syncthreads();
  __threadfence();
}

// ---------------- prep: f32->f16 weights, transpose x, zero state ----------------
__global__ void prep_kernel(const float* __restrict__ x,
                            const float* __restrict__ Whh0, const float* __restrict__ Wih1,
                            const float* __restrict__ Whh1, const float* __restrict__ Wdhh,
                            char* __restrict__ ws) {
  const int tid = blockIdx.x * blockDim.x + threadIdx.x;
  const int nth = gridDim.x * blockDim.x;
  _Float16* wh0 = (_Float16*)(ws + OFF_WH0);
  _Float16* wi1 = (_Float16*)(ws + OFF_WI1);
  _Float16* wh1 = (_Float16*)(ws + OFF_WH1);
  _Float16* wdh = (_Float16*)(ws + OFF_WDH);
  for (int i = tid; i < 3 * HD * HD; i += nth) {
    wh0[i] = (_Float16)Whh0[i];
    wi1[i] = (_Float16)Wih1[i];
    wh1[i] = (_Float16)Whh1[i];
    wdh[i] = (_Float16)Wdhh[i];
  }
  float* xT = (float*)(ws + OFF_XT);
  for (int i = tid; i < BN * LN; i += nth) {
    int b = i >> 9, t = i & (LN - 1);
    xT[t * BN + b] = x[i];                       // x is [B, 1, L]
  }
  _Float16* h0a = (_Float16*)(ws + OFF_H0A);
  _Float16* h1a = (_Float16*)(ws + OFF_H1A);
  for (int i = tid; i < BN * HD; i += nth) { h0a[i] = (_Float16)0.0f; h1a[i] = (_Float16)0.0f; }
  float* pred = (float*)(ws + OFF_PRED);
  for (int i = tid; i < BN; i += nth) pred[i] = 0.0f;
  if (tid == 0) *(unsigned*)(ws + OFF_BAR) = 0u;
}

// ---------------- persistent GRU scan: enc L0 -> enc L1 -> decoder ----------------
__global__ __launch_bounds__(512) void gru_scan(
    const float* __restrict__ wih0, const float* __restrict__ bih0, const float* __restrict__ bhh0,
    const float* __restrict__ bih1, const float* __restrict__ bhh1,
    const float* __restrict__ wdih, const float* __restrict__ bdih, const float* __restrict__ bdhh,
    const float* __restrict__ Wo,   const float* __restrict__ bo,
    float* __restrict__ out, char* __restrict__ ws) {
  const int lane  = threadIdx.x & 31;
  const int wid   = (blockIdx.x << 4) | (threadIdx.x >> 5);  // 0..255
  const int mbase = (wid >> 4) << 4;                         // batch tile base
  const int jbase = (wid & 15) << 4;                         // hidden tile base
  const int mloc  = (lane >> 4) << 3;                        // C-tile row group per lane
  const int j     = jbase + (lane & 15);                     // this lane's hidden column

  const _Float16* Wh0 = (const _Float16*)(ws + OFF_WH0);
  const _Float16* Wi1 = (const _Float16*)(ws + OFF_WI1);
  const _Float16* Wh1 = (const _Float16*)(ws + OFF_WH1);
  const _Float16* Wdh = (const _Float16*)(ws + OFF_WDH);
  const float*    xT  = (const float*)(ws + OFF_XT);
  _Float16* h0a = (_Float16*)(ws + OFF_H0A);
  _Float16* h0b = (_Float16*)(ws + OFF_H0B);
  _Float16* h1a = (_Float16*)(ws + OFF_H1A);
  _Float16* h1b = (_Float16*)(ws + OFF_H1B);
  _Float16* ys0 = (_Float16*)(ws + OFF_YS0);
  float*    pred = (float*)(ws + OFF_PRED);
  unsigned* bar  = (unsigned*)(ws + OFF_BAR);
  const unsigned NB = gridDim.x;
  unsigned epoch = 0;

  // ---------------- encoder layer 0 (input_size = 1) ----------------
  {
    const float wr = wih0[j], wz = wih0[HD + j], wn = wih0[2 * HD + j];
    const float bir = bih0[j], biz = bih0[HD + j], bin = bih0[2 * HD + j];
    const float bhr = bhh0[j], bhz = bhh0[HD + j], bhn = bhh0[2 * HD + j];
    for (int t = 0; t < LN; ++t) {
      const _Float16* hp = (t & 1) ? h0b : h0a;
      _Float16*       hc = (t & 1) ? h0a : h0b;
      v8f c0 = {}, c1 = {}, c2 = {};
      gemm3(hp, Wh0, mbase, jbase, lane, c0, c1, c2);
      const float* xt = xT + (size_t)t * BN;
      _Float16* ys = ys0 + (size_t)t * BN * HD;
#pragma unroll
      for (int r = 0; r < 8; ++r) {
        const int b2 = mbase + mloc + r;
        const float xv = xt[b2];
        const float rg = sigf(xv * wr + bir + c0[r] + bhr);
        const float zg = sigf(xv * wz + biz + c1[r] + bhz);
        const float ng = tanhf_fast(xv * wn + bin + rg * (c2[r] + bhn));
        const float hv = (1.0f - zg) * ng + zg * (float)hp[(size_t)b2 * HD + j];
        const _Float16 hh = (_Float16)hv;
        hc[(size_t)b2 * HD + j] = hh;
        ys[(size_t)b2 * HD + j] = hh;
      }
      grid_sync(bar, (++epoch) * NB);
    }
  }

  // ---------------- encoder layer 1 (input = ys0[t], fused gi+gh GEMMs) ----------------
  {
    const float bir = bih1[j], biz = bih1[HD + j], bin = bih1[2 * HD + j];
    const float bhr = bhh1[j], bhz = bhh1[HD + j], bhn = bhh1[2 * HD + j];
    for (int t = 0; t < LN; ++t) {
      const _Float16* hp = (t & 1) ? h1b : h1a;
      _Float16*       hc = (t & 1) ? h1a : h1b;
      const _Float16* ysp = ys0 + (size_t)t * BN * HD;
      v8f i0 = {}, i1 = {}, i2 = {}, g0 = {}, g1 = {}, g2 = {};
      gemm3(ysp, Wi1, mbase, jbase, lane, i0, i1, i2);   // input projection
      gemm3(hp,  Wh1, mbase, jbase, lane, g0, g1, g2);   // hidden projection
#pragma unroll
      for (int r = 0; r < 8; ++r) {
        const int b2 = mbase + mloc + r;
        const float rg = sigf(i0[r] + bir + g0[r] + bhr);
        const float zg = sigf(i1[r] + biz + g1[r] + bhz);
        const float ng = tanhf_fast(i2[r] + bin + rg * (g2[r] + bhn));
        const float hv = (1.0f - zg) * ng + zg * (float)hp[(size_t)b2 * HD + j];
        hc[(size_t)b2 * HD + j] = (_Float16)hv;
      }
      grid_sync(bar, (++epoch) * NB);
    }
  }

  // ---------------- decoder: 96 GRUCell steps + Linear(H,1) head ----------------
  {
    const float wr = wdih[j], wz = wdih[HD + j], wn = wdih[2 * HD + j];
    const float bir = bdih[j], biz = bdih[HD + j], bin = bdih[2 * HD + j];
    const float bhr = bdhh[j], bhz = bdhh[HD + j], bhn = bdhh[2 * HD + j];
    for (int t = 0; t < PL; ++t) {
      const _Float16* hp = (t & 1) ? h1b : h1a;   // continues from layer-1 final state (h1a)
      _Float16*       hc = (t & 1) ? h1a : h1b;
      v8f c0 = {}, c1 = {}, c2 = {};
      gemm3(hp, Wdh, mbase, jbase, lane, c0, c1, c2);
#pragma unroll
      for (int r = 0; r < 8; ++r) {
        const int b2 = mbase + mloc + r;
        const float xv = pred[b2];                // previous step's prediction (0 at t=0)
        const float rg = sigf(xv * wr + bir + c0[r] + bhr);
        const float zg = sigf(xv * wz + biz + c1[r] + bhz);
        const float ng = tanhf_fast(xv * wn + bin + rg * (c2[r] + bhn));
        const float hv = (1.0f - zg) * ng + zg * (float)hp[(size_t)b2 * HD + j];
        hc[(size_t)b2 * HD + j] = (_Float16)hv;
      }
      grid_sync(bar, (++epoch) * NB);
      // output head: 8 waves, one batch row per lane
      if (wid < 8) {
        const int b2 = (wid << 5) | lane;
        const _Float16* hb = hc + (size_t)b2 * HD;
        float acc = bo[0];
#pragma unroll 4
        for (int q = 0; q < HD; q += 8) {
          v8h hv = *(const v8h*)(hb + q);
#pragma unroll
          for (int e = 0; e < 8; ++e) acc += (float)hv[e] * Wo[q + e];
        }
        pred[b2] = acc;
        out[(size_t)b2 * PL + t] = acc;
      }
      grid_sync(bar, (++epoch) * NB);
    }
  }
}

extern "C" void kernel_launch(void* const* d_in, const int* in_sizes, int n_in,
                              void* d_out, int out_size, void* d_ws, size_t ws_size,
                              hipStream_t stream) {
  (void)in_sizes; (void)n_in; (void)out_size; (void)ws_size;
  const float* x    = (const float*)d_in[0];
  const float* Wih0 = (const float*)d_in[1];
  const float* Whh0 = (const float*)d_in[2];
  const float* bih0 = (const float*)d_in[3];
  const float* bhh0 = (const float*)d_in[4];
  const float* Wih1 = (const float*)d_in[5];
  const float* Whh1 = (const float*)d_in[6];
  const float* bih1 = (const float*)d_in[7];
  const float* bhh1 = (const float*)d_in[8];
  const float* Wdih = (const float*)d_in[9];
  const float* Wdhh = (const float*)d_in[10];
  const float* bdih = (const float*)d_in[11];
  const float* bdhh = (const float*)d_in[12];
  const float* Wo   = (const float*)d_in[13];
  const float* bo   = (const float*)d_in[14];
  char* ws = (char*)d_ws;

  prep_kernel<<<256, 256, 0, stream>>>(x, Whh0, Wih1, Whh1, Wdhh, ws);
  gru_scan<<<16, 512, 0, stream>>>(Wih0, bih0, bhh0, bih1, bhh1,
                                   Wdih, bdih, bdhh, Wo, bo,
                                   (float*)d_out, ws);
}
